// AttentionNet_223338300234
// MI455X (gfx1250) — matrix operand order repaired and checked by reading, not compiled, not run on hardware
//
#include <hip/hip_runtime.h>
#include <hip/hip_bf16.h>
#include <math.h>

// ---------------- problem constants (match reference) ----------------
#define NBLK  2
#define CCH   64
#define NH    4      // heads
#define FFD   256
#define NSEQ  40
#define SEQL  512
#define NPAIR 780
#define BB    2
#define NWAVES 4     // waves per block in WMMA kernels (blockDim = 128, wave32)

#ifndef __has_builtin
#define __has_builtin(x) 0
#endif
#if __has_builtin(__builtin_amdgcn_global_load_async_to_lds_b128) && \
    __has_builtin(__builtin_amdgcn_s_wait_asynccnt)
#define USE_ASYNC_LDS 1
#else
#define USE_ASYNC_LDS 0
#endif

typedef __attribute__((ext_vector_type(16))) _Float16 v16h;
typedef __attribute__((ext_vector_type(8)))  _Float16 v8h;
typedef __attribute__((ext_vector_type(8)))  float    v8f;

static __device__ __forceinline__ v8f wmma32(v16h a, v16h b, v8f c) {
  // D = A(16x32 f16) * B(32x16 f16) + C(16x16 f32)
  return __builtin_amdgcn_wmma_f32_16x16x32_f16(false, a, false, b, (short)0, c,
                                                false, false);
}

static __device__ __forceinline__ v16h cat8(v8h lo, v8h hi) {
  return __builtin_shufflevector(lo, hi, 0,1,2,3,4,5,6,7,8,9,10,11,12,13,14,15);
}

// A fragment (16x32 f16, ISA layout): row m = lane&15; lane-group h = lane>>4
// selects K chunks {32s+8h .. +7} and {32s+16+8h .. +7} from a row-major LDS
// tile with row length `ld` halves.
static __device__ __forceinline__ v16h afrag(const _Float16* t, int ld, int m,
                                             int s, int h) {
  const _Float16* p = t + m * ld + s * 32 + h * 8;
  v8h lo = *(const v8h*)p;
  v8h hi = *(const v8h*)(p + 16);
  return cat8(lo, hi);
}

// B fragment (32x16 f16): col n = 16*nt + (lane&15); K chunks {32s+8h .. +7}
// and {32s+16+8h .. +7}, h = lane>>4. Weights stored row-major [N][K] (w[n][k]),
// which is exactly y = x @ w.T.
static __device__ __forceinline__ v16h bfrag(const _Float16* w, int ldk, int nt,
                                             int s, int lane) {
  const _Float16* p = w + (nt * 16 + (lane & 15)) * ldk + s * 32 + (lane >> 4) * 8;
  v8h lo = *(const v8h*)p;
  v8h hi = *(const v8h*)(p + 16);
  return cat8(lo, hi);
}

// intra-wave LDS producer->consumer ordering (lowers to s_wait_dscnt)
static __device__ __forceinline__ void wave_sync_lds() {
  __builtin_amdgcn_fence(__ATOMIC_RELEASE, "workgroup");
  __builtin_amdgcn_wave_barrier();
  __builtin_amdgcn_fence(__ATOMIC_ACQUIRE, "workgroup");
}

#if USE_ASYNC_LDS
// Async DMA of 16B global -> LDS (tracked by ASYNCcnt, overlaps with WMMA).
// ROCm 7.2 clang-22 signature: (v4i addrspace(1)*, v4i addrspace(3)*, Ii, Ii)
typedef int v4i __attribute__((vector_size(16)));
static __device__ __forceinline__ void async_ld16(const float* g, float* l) {
  __builtin_amdgcn_global_load_async_to_lds_b128(
      (__attribute__((address_space(1))) v4i*)g,
      (__attribute__((address_space(3))) v4i*)l, 0, 0);
}
#endif

static __device__ __forceinline__ float elu1(float x) {
  return (x > 0.f) ? (x + 1.f) : __expf(x);   // elu(x)+1
}

// Stage one 16-token x tile (f32 global, token stride strideT) into f16 LDS
// tile xt[16][64], optionally via async-to-LDS through f32 staging xs.
static __device__ __forceinline__ void stage_xtile(
    const float* __restrict__ hp, long long base, long long strideT, int tok0,
    int Tdim, float* xs, _Float16* xt, int lane) {
#if USE_ASYNC_LDS
  for (int i = lane; i < 256; i += 32) {         // 256 x float4 chunks
    int e = i * 4;
    int tok = tok0 + (e >> 6);
    int c = e & 63;
    if (tok < Tdim) {
      async_ld16(hp + base + (long long)tok * strideT + c, xs + e);
    } else {
      xs[e] = 0.f; xs[e + 1] = 0.f; xs[e + 2] = 0.f; xs[e + 3] = 0.f;
    }
  }
  __builtin_amdgcn_s_wait_asynccnt(0);
  wave_sync_lds();
  for (int i = lane; i < 16 * CCH; i += 32) xt[i] = (_Float16)xs[i];
#else
  (void)xs;
  for (int i = lane; i < 16 * CCH; i += 32) {
    int tok = tok0 + (i >> 6);
    int c = i & 63;
    float v = (tok < Tdim) ? hp[base + (long long)tok * strideT + c] : 0.f;
    xt[i] = (_Float16)v;
  }
#endif
  wave_sync_lds();
}

// ---------------- input conv 22->64 + ReLU ----------------
__global__ void conv_in_kernel(const float* __restrict__ x,
                               const float* __restrict__ w,
                               const float* __restrict__ bias,
                               float* __restrict__ h) {
  int idx = blockIdx.x * blockDim.x + threadIdx.x;
  const int total = BB * SEQL * NSEQ;
  if (idx >= total) return;
  int s = idx % NSEQ;
  int l = (idx / NSEQ) % SEQL;
  int b = idx / (NSEQ * SEQL);
  float xv[22];
#pragma unroll
  for (int c = 0; c < 22; ++c)
    xv[c] = x[(((size_t)b * 22 + c) * SEQL + l) * NSEQ + s];
  float* out = h + (size_t)idx * CCH;
  for (int d = 0; d < CCH; ++d) {
    float a = bias[d];
#pragma unroll
    for (int c = 0; c < 22; ++c) a += w[d * 22 + c] * xv[c];
    out[d] = fmaxf(a, 0.f);
  }
}

// ---------------- seq2pair gather-sum + LayerNorm ----------------
__global__ void gather_ln_kernel(const float* __restrict__ h,
                                 const float* __restrict__ g0,
                                 const float* __restrict__ be0,
                                 float* __restrict__ hp) {
  int idx = blockIdx.x * blockDim.x + threadIdx.x;
  const int total = BB * NPAIR * SEQL;
  if (idx >= total) return;
  int l = idx % SEQL;
  int p = (idx / SEQL) % NPAIR;
  int b = idx / (SEQL * NPAIR);
  // decode np.triu_indices(NSEQ,1) order
  int i = 0, rem = p;
  while (rem >= NSEQ - 1 - i) { rem -= NSEQ - 1 - i; ++i; }
  int j = i + 1 + rem;
  const float* hi_ = h + (((size_t)b * SEQL + l) * NSEQ + i) * CCH;
  const float* hj_ = h + (((size_t)b * SEQL + l) * NSEQ + j) * CCH;
  float v[CCH];
  float mu = 0.f, m2 = 0.f;
#pragma unroll
  for (int c = 0; c < CCH; ++c) {
    float t = hi_[c] + hj_[c];
    v[c] = t; mu += t; m2 += t * t;
  }
  mu *= (1.f / CCH);
  m2 = m2 * (1.f / CCH) - mu * mu;
  float rs = rsqrtf(m2 + 1e-5f);
  float* out = hp + (size_t)idx * CCH;
#pragma unroll
  for (int c = 0; c < CCH; ++c) out[c] = (v[c] - mu) * rs * g0[c] + be0[c];
}

// ---------------- fused linear-attention (row or column axis) ----------------
// grid.x = B * Mdim; one workgroup owns one (b,m) sequence of Tdim tokens.
// Writes hp += proj(attention) in place (exclusive ownership per workgroup).
__global__ __launch_bounds__(128) void attn_kernel(
    float* __restrict__ hp,
    const float* __restrict__ wq, const float* __restrict__ bq,
    const float* __restrict__ wk, const float* __restrict__ bk,
    const float* __restrict__ wv, const float* __restrict__ bv,
    const float* __restrict__ wp, const float* __restrict__ bp,
    int Mdim, int Tdim, long long strideM, long long strideT, int ntiles) {
  extern __shared__ char smem[];
  _Float16* Wq = (_Float16*)smem;          // 64*64 f16 each
  _Float16* Wk = Wq + 4096;
  _Float16* Wv = Wk + 4096;
  _Float16* Wp = Wv + 4096;
  float* Bq = (float*)(Wp + 4096);
  float* Bk = Bq + CCH;
  float* Bv = Bk + CCH;
  float* Bp = Bv + CCH;
  float* KS  = Bp + CCH;                   // ksum[64]
  float* KTV = KS + CCH;                   // ktv_cs[64] (head-major: c = 16*head+e)
  _Float16* QB = (_Float16*)(KTV + CCH);   // q buffer [ntiles*16][64] f16
  _Float16* XT = QB + (size_t)ntiles * 16 * CCH;   // per-wave x tiles f16
  float* XS = (float*)(XT + NWAVES * 16 * CCH);    // per-wave f32 async stage

  const int tid = threadIdx.x;
  const int wave = tid >> 5, lane = tid & 31;
  const int mrel = lane & 15, hgrp = lane >> 4;

  const int bidx = blockIdx.x / Mdim;
  const int m    = blockIdx.x % Mdim;
  const long long base =
      (long long)bidx * (long long)NPAIR * SEQL * CCH + (long long)m * strideM;

  if (tid == 0) {
    __builtin_prefetch(wq, 0, 1);
    __builtin_prefetch(wk, 0, 1);
    __builtin_prefetch(wv, 0, 1);
    __builtin_prefetch(wp, 0, 1);
  }
  for (int i = tid; i < 4096; i += 128) {
    Wq[i] = (_Float16)wq[i];
    Wk[i] = (_Float16)wk[i];
    Wv[i] = (_Float16)wv[i];
    Wp[i] = (_Float16)wp[i];
  }
  for (int i = tid; i < CCH; i += 128) {
    Bq[i] = bq[i]; Bk[i] = bk[i]; Bv[i] = bv[i]; Bp[i] = bp[i];
    KS[i] = 0.f; KTV[i] = 0.f;
  }
  __syncthreads();

  float ksloc[NH]  = {0.f, 0.f, 0.f, 0.f};
  float ktvloc[NH] = {0.f, 0.f, 0.f, 0.f};
  _Float16* xt = XT + wave * (16 * CCH);
  float* xs = XS + wave * (16 * CCH);

  // ---------- pass 1: q,k,v projections; ksum and ktv_cs reductions ----------
  for (int tile = wave; tile < ntiles; tile += NWAVES) {
    stage_xtile(hp, base, strideT, tile * 16, Tdim, xs, xt, lane);

    v16h a0 = afrag(xt, CCH, mrel, 0, hgrp);
    v16h a1 = afrag(xt, CCH, mrel, 1, hgrp);

    float vfr[NH][8];
#pragma unroll
    for (int nt = 0; nt < NH; ++nt) {             // v = xWv^T + bv (one N-tile == one head)
      v8f acc = {};
      acc = wmma32(a0, bfrag(Wv, CCH, nt, 0, lane), acc);
      acc = wmma32(a1, bfrag(Wv, CCH, nt, 1, lane), acc);
      float bias = Bv[nt * 16 + mrel];
#pragma unroll
      for (int r = 0; r < 8; ++r) vfr[nt][r] = acc[r] + bias;
    }
#pragma unroll
    for (int nt = 0; nt < NH; ++nt) {             // k = elu(xWk^T+bk)+1
      v8f acc = {};
      acc = wmma32(a0, bfrag(Wk, CCH, nt, 0, lane), acc);
      acc = wmma32(a1, bfrag(Wk, CCH, nt, 1, lane), acc);
      float bias = Bk[nt * 16 + mrel];
#pragma unroll
      for (int r = 0; r < 8; ++r) {
        int tok = tile * 16 + r + 8 * hgrp;
        float kk = (tok < Tdim) ? elu1(acc[r] + bias) : 0.f;
        ksloc[nt] += kk;                          // ksum partial
        float kb = kk;                            // kbar[t] = sum_d k[t,d]
        kb += __shfl_xor(kb, 1);
        kb += __shfl_xor(kb, 2);
        kb += __shfl_xor(kb, 4);
        kb += __shfl_xor(kb, 8);
        ktvloc[nt] += kb * ((tok < Tdim) ? vfr[nt][r] : 0.f);  // ktv_cs partial
      }
    }
#pragma unroll
    for (int nt = 0; nt < NH; ++nt) {             // q = elu(xWq^T+bq)+1 -> LDS
      v8f acc = {};
      acc = wmma32(a0, bfrag(Wq, CCH, nt, 0, lane), acc);
      acc = wmma32(a1, bfrag(Wq, CCH, nt, 1, lane), acc);
      float bias = Bq[nt * 16 + mrel];
#pragma unroll
      for (int r = 0; r < 8; ++r) {
        int tok = tile * 16 + r + 8 * hgrp;
        if (tok < Tdim)
          QB[(size_t)tok * CCH + nt * 16 + mrel] = (_Float16)elu1(acc[r] + bias);
      }
    }
  }
#pragma unroll
  for (int nt = 0; nt < NH; ++nt) {
    atomicAdd(&KS[nt * 16 + mrel], ksloc[nt]);    // ds_add_f32
    atomicAdd(&KTV[nt * 16 + mrel], ktvloc[nt]);
  }
  __syncthreads();

  // ---------- pass 2: z = 1/(q.ksum+eps); out = (z*ktv) Wp^T + bp + residual ----------
  for (int tile = wave; tile < ntiles; tile += NWAVES) {
    int tokm = tile * 16 + mrel;
    float z[NH];
    if (tokm < Tdim) {
#pragma unroll
      for (int n = 0; n < NH; ++n) {
        float s = 0.f;
#pragma unroll
        for (int d = 0; d < 16; ++d)
          s += (float)QB[(size_t)tokm * CCH + n * 16 + d] * KS[n * 16 + d];
        z[n] = 1.f / (s + 1e-6f);
      }
    } else {
#pragma unroll
      for (int n = 0; n < NH; ++n) z[n] = 0.f;
    }
    // synthesize attention A-fragments directly (att[m][c] = z[m][c>>4]*ktv[c])
    v16h a0, a1;
#pragma unroll
    for (int e = 0; e < 16; ++e) {
      int K0 = (e < 8) ? (8 * hgrp + e) : (16 + 8 * hgrp + (e - 8));
      int K1 = 32 + K0;
      a0[e] = (_Float16)(z[K0 >> 4] * KTV[K0]);
      a1[e] = (_Float16)(z[K1 >> 4] * KTV[K1]);
    }
#pragma unroll
    for (int nt = 0; nt < NH; ++nt) {
      v8f acc = {};
      acc = wmma32(a0, bfrag(Wp, CCH, nt, 0, lane), acc);
      acc = wmma32(a1, bfrag(Wp, CCH, nt, 1, lane), acc);
      int n = nt * 16 + mrel;
      float bias = Bp[n];
#pragma unroll
      for (int r = 0; r < 8; ++r) {
        int tok = tile * 16 + r + 8 * hgrp;
        if (tok < Tdim) {
          long long addr = base + (long long)tok * strideT + n;
          hp[addr] += acc[r] + bias;              // residual fused
        }
      }
    }
  }
}

// ---------------- LayerNorm over channels ----------------
__global__ void ln_kernel(float* __restrict__ hp, const float* __restrict__ g,
                          const float* __restrict__ b, int ntok) {
  int t = blockIdx.x * blockDim.x + threadIdx.x;
  if (t >= ntok) return;
  float* p = hp + (size_t)t * CCH;
  float mu = 0.f, m2 = 0.f;
#pragma unroll
  for (int c = 0; c < CCH; ++c) { float v = p[c]; mu += v; m2 += v * v; }
  mu *= (1.f / CCH);
  m2 = m2 * (1.f / CCH) - mu * mu;
  float rs = rsqrtf(m2 + 1e-5f);
#pragma unroll
  for (int c = 0; c < CCH; ++c) p[c] = (p[c] - mu) * rs * g[c] + b[c];
}

// ---------------- FFN 64 -> 256 (GELU) -> 64, residual ----------------
__global__ __launch_bounds__(128) void ffn_kernel(
    float* __restrict__ hp, const float* __restrict__ w1,
    const float* __restrict__ b1, const float* __restrict__ w2,
    const float* __restrict__ b2, int ntiles) {
  extern __shared__ char smem[];
  _Float16* W1 = (_Float16*)smem;              // [256][64] f16
  _Float16* W2 = W1 + FFD * CCH;               // [64][256] f16
  float* B1 = (float*)(W2 + CCH * FFD);
  float* B2 = B1 + FFD;
  _Float16* XT = (_Float16*)(B2 + CCH);        // 4 waves * [16][64] f16
  _Float16* H1 = XT + NWAVES * 16 * CCH;       // 4 waves * [16][256] f16
  float* XS = (float*)(H1 + NWAVES * 16 * FFD); // 4 waves * [16][64] f32 stage

  const int tid = threadIdx.x, wave = tid >> 5, lane = tid & 31;
  const int mrel = lane & 15, hgrp = lane >> 4;

  if (tid == 0) { __builtin_prefetch(w1, 0, 1); __builtin_prefetch(w2, 0, 1); }
  for (int i = tid; i < FFD * CCH; i += 128) {
    W1[i] = (_Float16)w1[i];
    W2[i] = (_Float16)w2[i];
  }
  for (int i = tid; i < FFD; i += 128) B1[i] = b1[i];
  for (int i = tid; i < CCH; i += 128) B2[i] = b2[i];
  __syncthreads();

  int tile = blockIdx.x * NWAVES + wave;
  if (tile >= ntiles) return;
  long long tbase = (long long)tile * 16 * CCH;

  _Float16* xt = XT + wave * (16 * CCH);
  _Float16* h1 = H1 + wave * (16 * FFD);
  float* xs = XS + wave * (16 * CCH);
  stage_xtile(hp, tbase, (long long)CCH, 0, 16, xs, xt, lane);

  v16h a0 = afrag(xt, CCH, mrel, 0, hgrp);
  v16h a1 = afrag(xt, CCH, mrel, 1, hgrp);

#pragma unroll
  for (int nt = 0; nt < FFD / 16; ++nt) {        // h1 = gelu(x W1^T + b1)
    v8f acc = {};
    acc = wmma32(a0, bfrag(W1, CCH, nt, 0, lane), acc);
    acc = wmma32(a1, bfrag(W1, CCH, nt, 1, lane), acc);
    int n = nt * 16 + mrel;
    float bias = B1[n];
#pragma unroll
    for (int r = 0; r < 8; ++r) {
      float xg = acc[r] + bias;
      float gl = 0.5f * xg * (1.f + erff(xg * 0.70710678118f));  // exact gelu
      h1[(r + 8 * hgrp) * FFD + n] = (_Float16)gl;
    }
  }
  wave_sync_lds();

#pragma unroll
  for (int nt = 0; nt < CCH / 16; ++nt) {        // y = h1 W2^T + b2 + x
    v8f acc = {};
#pragma unroll
    for (int s = 0; s < FFD / 32; ++s) {
      v16h a = afrag(h1, FFD, mrel, s, hgrp);
      acc = wmma32(a, bfrag(W2, FFD, nt, s, lane), acc);
    }
    int n = nt * 16 + mrel;
    float bias = B2[n];
#pragma unroll
    for (int r = 0; r < 8; ++r) {
      long long addr = tbase + (long long)(r + 8 * hgrp) * CCH + n;
      hp[addr] += acc[r] + bias;
    }
  }
}

// ---------------- head: softplus(hp.wout+b), mean over L ----------------
__global__ __launch_bounds__(32) void head_kernel(const float* __restrict__ hp,
                                                  const float* __restrict__ wout,
                                                  const float* __restrict__ bout,
                                                  float* __restrict__ out) {
  int bp = blockIdx.x;          // 0 .. B*P-1
  int lane = threadIdx.x;
  float acc = 0.f;
  for (int l = lane; l < SEQL; l += 32) {
    const float* p = hp + ((size_t)bp * SEQL + l) * CCH;
    float d = bout[0];
#pragma unroll
    for (int c = 0; c < CCH; ++c) d += p[c] * wout[c];
    acc += (d > 20.f) ? d : log1pf(__expf(d));   // softplus
  }
  acc += __shfl_xor(acc, 1);
  acc += __shfl_xor(acc, 2);
  acc += __shfl_xor(acc, 4);
  acc += __shfl_xor(acc, 8);
  acc += __shfl_xor(acc, 16);
  if (lane == 0) out[bp] = acc * (1.f / SEQL);
}

// ---------------- launch ----------------
extern "C" void kernel_launch(void* const* d_in, const int* in_sizes, int n_in,
                              void* d_out, int out_size, void* d_ws,
                              size_t ws_size, hipStream_t stream) {
  (void)in_sizes; (void)n_in; (void)out_size; (void)ws_size;
  const float* x    = (const float*)d_in[0];
  const float* w_in = (const float*)d_in[1];
  const float* b_in = (const float*)d_in[2];
  const float* g0   = (const float*)d_in[3];
  const float* be0  = (const float*)d_in[4];
  const float* rqw  = (const float*)d_in[5];
  const float* rqb  = (const float*)d_in[6];
  const float* rkw  = (const float*)d_in[7];
  const float* rkb  = (const float*)d_in[8];
  const float* rvw  = (const float*)d_in[9];
  const float* rvb  = (const float*)d_in[10];
  const float* rpw  = (const float*)d_in[11];
  const float* rpb  = (const float*)d_in[12];
  const float* cqw  = (const float*)d_in[13];
  const float* cqb  = (const float*)d_in[14];
  const float* ckw  = (const float*)d_in[15];
  const float* ckb  = (const float*)d_in[16];
  const float* cvw  = (const float*)d_in[17];
  const float* cvb  = (const float*)d_in[18];
  const float* cpw  = (const float*)d_in[19];
  const float* cpb  = (const float*)d_in[20];
  const float* ln_g = (const float*)d_in[21];
  const float* ln_b = (const float*)d_in[22];
  const float* f1w  = (const float*)d_in[23];
  const float* f1b  = (const float*)d_in[24];
  const float* f2w  = (const float*)d_in[25];
  const float* f2b  = (const float*)d_in[26];
  const float* wout = (const float*)d_in[27];
  const float* bout = (const float*)d_in[28];
  // d_in[29] = seq2pair: decoded arithmetically in gather_ln_kernel.

  float* hbuf = (float*)d_ws;                              // [B,L,S,C] f32
  float* hp = hbuf + (size_t)BB * SEQL * NSEQ * CCH;       // [B,P,L,C] f32

  const int ntok = BB * NPAIR * SEQL;                      // 798720
  {
    int n = BB * SEQL * NSEQ;
    conv_in_kernel<<<(n + 255) / 256, 256, 0, stream>>>(x, w_in, b_in, hbuf);
  }
  gather_ln_kernel<<<(ntok + 255) / 256, 256, 0, stream>>>(hbuf, g0, be0, hp);

  const int ntR = SEQL / 16;                 // 32 token-tiles (row axis, T=L)
  const int ntC = (NPAIR + 15) / 16;         // 49 token-tiles (col axis, T=P)
  // layout must match attn_kernel: W(4x8KB) + B(1KB) + KS/KTV(0.5KB) + QB + XT + XS
  auto attn_smem = [](int nt) {
    return (size_t)(4 * 4096 * 2 + 4 * CCH * 4 + 2 * CCH * 4 +
                    (size_t)nt * 16 * CCH * 2 + NWAVES * 16 * CCH * 2 +
                    NWAVES * 16 * CCH * 4);
  };
  const size_t smR = attn_smem(ntR);         // ~124 KB
  const size_t smC = attn_smem(ntC);         // ~156 KB (<= 320 KB/WG)
  const size_t smF = (size_t)2 * FFD * CCH * 2 + FFD * 4 + CCH * 4 +
                     NWAVES * 16 * CCH * 2 + NWAVES * 16 * FFD * 2 +
                     NWAVES * 16 * CCH * 4;  // ~121 KB

  for (int blk = 0; blk < NBLK; ++blk) {
    // row attention: M = P, T = L (tokens contiguous: strideT = C)
    attn_kernel<<<BB * NPAIR, 128, smR, stream>>>(
        hp, rqw + blk * 4096, rqb + blk * 64, rkw + blk * 4096, rkb + blk * 64,
        rvw + blk * 4096, rvb + blk * 64, rpw + blk * 4096, rpb + blk * 64,
        NPAIR, SEQL, (long long)SEQL * CCH, (long long)CCH, ntR);
    ln_kernel<<<(ntok + 255) / 256, 256, 0, stream>>>(hp, ln_g + blk * 64,
                                                      ln_b + blk * 64, ntok);
    // column attention: M = L, T = P (strideT = L*C)
    attn_kernel<<<BB * SEQL, 128, smC, stream>>>(
        hp, cqw + blk * 4096, cqb + blk * 64, ckw + blk * 4096, ckb + blk * 64,
        cvw + blk * 4096, cvb + blk * 64, cpw + blk * 4096, cpb + blk * 64,
        SEQL, NPAIR, (long long)CCH, (long long)SEQL * CCH, ntC);
    ln_kernel<<<(ntok + 255) / 256, 256, 0, stream>>>(hp, ln_g + blk * 64,
                                                      ln_b + blk * 64, ntok);
    // FFN
    int ftiles = ntok / 16;                  // 49920, exact
    ffn_kernel<<<(ftiles + NWAVES - 1) / NWAVES, 128, smF, stream>>>(
        hp, f1w + blk * FFD * CCH, f1b + blk * FFD, f2w + blk * CCH * FFD,
        f2b + blk * CCH, ftiles);
    if (blk != NBLK - 1)
      ln_kernel<<<(ntok + 255) / 256, 256, 0, stream>>>(hp, ln_g + blk * 64,
                                                        ln_b + blk * 64, ntok);
  }
  head_kernel<<<BB * NPAIR, 32, 0, stream>>>(hp, wout, bout, (float*)d_out);
}
